// DVAEdgeEncoder_82068235092594
// MI455X (gfx1250) — matrix operand order repaired and checked by reading, not compiled, not run on hardware
//
#include <hip/hip_runtime.h>
#include <hip/hip_bf16.h>

// ---------------------------------------------------------------------------
// D-VAE DAG GNN encoder for MI455X (gfx1250), compile-only tuned.
// Single persistent workgroup (8 wave32 waves), bf16 WMMA GEMMs,
// async-to-LDS weight staging (ASYNCcnt path), LDS strip staging.
// ---------------------------------------------------------------------------

#define Bsz 32
#define Nv  24
#define HS  128
#define NROWS (Nv * Bsz)       // 768 rows, row = j*32 + b
#define STRIDE_S 130           // padded LDS strip stride (bank-conflict free)
#define STRIDE_W 136           // padded LDS weight stride (4-bank skew, 16B aligned)

typedef __attribute__((ext_vector_type(16))) __bf16 v16bf;
typedef __attribute__((ext_vector_type(8)))  float  v8f;
typedef int v4i_g __attribute__((vector_size(16)));   // matches builtin param type

union FragAB { unsigned u[8]; v16bf v; };
union FragC  { float    f[8]; v8f   v; };

#if defined(__has_builtin)
#if __has_builtin(__builtin_amdgcn_global_load_async_to_lds_b128)
#define HAVE_ASYNC_LDS 1
#endif
#endif

// bf16 weight segments inside d_ws (element offsets)
#define OFF_WHH_FE 0
#define OFF_WHH_BE 49152
#define OFF_WHH_FV 98304
#define OFF_WHH_BV 147456
#define OFF_WG_F   196608
#define OFF_WM_F   212992
#define OFF_WG_B   229376
#define OFF_WM_B   245760
#define WB_ELEMS   262144

// byte offsets inside d_ws
#define WS_WB     0u
#define WS_HST    524288u      // bf16 [768][128]
#define WS_CARRYF 720896u      // f32 [32][128]
#define WS_CARRYB 737280u      // f32 [32][128]
#define WS_SFFIN  753664u      // f32 [32][128]

static __device__ __forceinline__ unsigned short f2bf(float f) {
  unsigned u = __builtin_bit_cast(unsigned, f);
  u += 0x7FFFu + ((u >> 16) & 1u);          // round-to-nearest-even
  return (unsigned short)(u >> 16);
}
static __device__ __forceinline__ float bf2f(unsigned short h) {
  return __builtin_bit_cast(float, ((unsigned)h) << 16);
}
static __device__ __forceinline__ float sigmoidf(float x) {
  return 1.0f / (1.0f + __expf(-x));
}
static __device__ __forceinline__ float tanh_fast(float x) {
  return 1.0f - 2.0f / (__expf(2.0f * x) + 1.0f);
}

// A-fragment (16x32 bf16): lane holds row m = lane&15; per ISA 7.12.2:
//   k = k0 + (r<4?0:16) + 8*(lane>=16) + 2*(r&3)
static __device__ __forceinline__ void loadA(FragAB& A, const unsigned short* base,
                                             int row0, int stride, int k0, int lane) {
  const int m    = lane & 15;
  const int koff = (lane >> 4) << 3;
  const unsigned short* p = base + (row0 + m) * stride;
#pragma unroll
  for (int r = 0; r < 8; ++r) {
    const int k = k0 + ((r < 4) ? 0 : 16) + koff + ((r & 3) << 1);
    A.u[r] = *(const unsigned*)(p + k);
  }
}

// B-fragment (32x16 bf16) built from row-major weight W[nrows][stride] with
// B[k][n] = W[n0+n][k]. lane holds col n = lane&15; k = k0 + (r<4?0:16) + 8*(lane>=16) + 2*(r&3).
// The 8 dwords are two runs of 16 contiguous bytes -> compiler fuses into 2x b128 loads.
static __device__ __forceinline__ void loadB(FragAB& Bf, const unsigned short* base,
                                             int n0, int stride, int k0, int lane) {
  const int n    = lane & 15;
  const int koff = (lane >> 4) << 3;
  const unsigned short* p = base + (n0 + n) * stride;
#pragma unroll
  for (int r = 0; r < 8; ++r) {
    const int k = k0 + ((r < 4) ? 0 : 16) + koff + ((r & 3) << 1);
    Bf.u[r] = *(const unsigned*)(p + k);
  }
}

static __device__ __forceinline__ v8f wmma_bf16(const FragAB& a, const FragAB& b, v8f c) {
  return __builtin_amdgcn_wmma_f32_16x16x32_bf16(false, a.v, false, b.v, (short)0, c,
                                                 false, false);
}

// ---------------------------------------------------------------------------
// prep: convert the 8 big weight matrices to bf16 (K-major rows of 128)
// ---------------------------------------------------------------------------
__global__ void dvae_prep_kernel(const float* whh_fe, const float* whh_be,
                                 const float* whh_fv, const float* whh_bv,
                                 const float* gate_f_W, const float* map_f_W,
                                 const float* gate_b_W, const float* map_b_W,
                                 unsigned short* wb) {
  int i = blockIdx.x * 256 + threadIdx.x;
  if (i >= WB_ELEMS) return;
  float val;
  if (i < OFF_WHH_BE)      val = whh_fe[i];
  else if (i < OFF_WHH_FV) val = whh_be[i - OFF_WHH_BE];
  else if (i < OFF_WHH_BV) val = whh_fv[i - OFF_WHH_FV];
  else if (i < OFF_WG_F)   val = whh_bv[i - OFF_WHH_BV];
  else {
    int l = i - OFF_WG_F;
    int seg = l >> 14, li = l & 16383;
    int r = li >> 7, c = li & 127;
    const float* src = (seg == 0) ? gate_f_W : (seg == 1) ? map_f_W
                      : (seg == 2) ? gate_b_W : map_b_W;
    val = src[r * 131 + c];                 // drop the 3 one-hot columns (handled f32)
  }
  wb[i] = f2bf(val);
}

// ---------------------------------------------------------------------------
struct KParams {
  const int* v_types; const int* e_types; const float* adj;
  const float* Wih_fv; const float* bih_fv; const float* bhh_fv;
  const float* Wih_bv; const float* bih_bv; const float* bhh_bv;
  const float* Wih_fe; const float* bih_fe; const float* bhh_fe;
  const float* Wih_be; const float* bih_be; const float* bhh_be;
  const float* gate_f_W; const float* gate_f_b; const float* map_f_W;
  const float* gate_b_W; const float* gate_b_b; const float* map_b_W;
  const float* unify_W; const float* unify_b; const float* ubn_g; const float* ubn_b;
  const float* cls_W1; const float* cls_b1; const float* cbn_g; const float* cbn_b;
  const float* cls_W2; const float* cls_b2;
  const unsigned short* wb;
  unsigned short* Hst;          // bf16 [768][128] recurrent state, L2-resident
  float* carryF; float* carryB; // H0 carries per direction
  float* SfFin;                 // Sf[:, N-1] snapshot after last forward pass
  float* out;
};

__global__ __launch_bounds__(256) void dvae_persistent_kernel(KParams P) {
  __shared__ __align__(16) unsigned short s_whh[384 * STRIDE_W];     // staged edge Whh (bf16)
  __shared__ __align__(16) unsigned short s_strip[8][16 * STRIDE_S]; // per-wave He strips
  __shared__ float s_Hmsg[Bsz * HS];                                 // message accumulator

  const int tid  = threadIdx.x;
  const int lane = tid & 31;
  const int wave = tid >> 5;

  // zero the H0 carries (deterministic across graph replays)
  for (int i = tid; i < Bsz * HS; i += 256) { P.carryF[i] = 0.f; P.carryB[i] = 0.f; }
  __threadfence();
  __syncthreads();

  for (int pass = 0; pass < 6; ++pass) {
    const bool rev = (pass & 1);
    const unsigned short* WhhE = P.wb + (rev ? OFF_WHH_BE : OFF_WHH_FE);
    const unsigned short* WhhV = P.wb + (rev ? OFF_WHH_BV : OFF_WHH_FV);
    const unsigned short* WgB  = P.wb + (rev ? OFF_WG_B : OFF_WG_F);
    const unsigned short* WmB  = P.wb + (rev ? OFF_WM_B : OFF_WM_F);
    const float* WihE = rev ? P.Wih_be : P.Wih_fe;
    const float* bihE = rev ? P.bih_be : P.bih_fe;
    const float* bhhE = rev ? P.bhh_be : P.bhh_fe;
    const float* WihV = rev ? P.Wih_bv : P.Wih_fv;
    const float* bihV = rev ? P.bih_bv : P.bih_fv;
    const float* bhhV = rev ? P.bhh_bv : P.bhh_fv;
    const float* Wg32 = rev ? P.gate_b_W : P.gate_f_W;   // [128][131]
    const float* Wm32 = rev ? P.map_b_W : P.map_f_W;     // [128][131]
    const float* bgF  = rev ? P.gate_b_b : P.gate_f_b;
    float* carry = rev ? P.carryB : P.carryF;
    const int start = rev ? (Nv - 1) : 0;

    // ---- stage this direction's edge-GRU Whh into LDS (async copy path) ----
    for (int c = tid; c < 384 * 16; c += 256) {
      const int row = c >> 4, cc = c & 15;                  // 16-byte chunks
      const unsigned short* src = WhhE + row * HS + cc * 8;
      unsigned short* dst = s_whh + row * STRIDE_W + cc * 8;
#ifdef HAVE_ASYNC_LDS
      __builtin_amdgcn_global_load_async_to_lds_b128(
          (v4i_g*)src,
          (__attribute__((address_space(3))) v4i_g*)dst, 0, 0);
#else
      *(uint4*)dst = *(const uint4*)src;
#endif
    }
#ifdef HAVE_ASYNC_LDS
    asm volatile("s_wait_asynccnt 0" ::: "memory");
#endif

    // H_states = zeros at pass start
    {
      unsigned* Hz = (unsigned*)P.Hst;
      for (int i = tid; i < NROWS * HS / 2; i += 256) Hz[i] = 0u;
    }
    __threadfence();
    __syncthreads();

    for (int s = 0; s < Nv; ++s) {
      const int v = rev ? (Nv - 1 - s) : s;

      for (int i = tid; i < Bsz * HS; i += 256) s_Hmsg[i] = 0.f;
      __syncthreads();

      // -------- Phase 1+2: per 16-row strip, fused edge-GRU + gate/map ------
      unsigned short* strip = s_strip[wave];
      for (int st = wave; st < NROWS / 16; st += 8) {
        const int row0 = st * 16;
        if (st + 8 < NROWS / 16)
          __builtin_prefetch(P.Hst + (st + 8) * 16 * HS, 0, 1);

        int   et[8];
        float msk[8];
#pragma unroll
        for (int r = 0; r < 8; ++r) {
          const int m = r + ((lane >> 4) << 3);
          const int R = row0 + m, j = R >> 5, b = R & 31;
          const int eidx = rev ? (b * (Nv * Nv) + v * Nv + j)
                               : (b * (Nv * Nv) + j * Nv + v);
          et[r]  = P.e_types[eidx];
          msk[r] = P.adj[eidx];
        }

        FragAB A[4];
#pragma unroll
        for (int kc = 0; kc < 4; ++kc) loadA(A[kc], P.Hst, row0, HS, kc * 32, lane);

        for (int ct = 0; ct < 8; ++ct) {
          const int h = ct * 16 + (lane & 15);
          FragC Cr, Cz, Cn;
          Cr.v = (v8f){0,0,0,0,0,0,0,0}; Cz.v = Cr.v; Cn.v = Cr.v;
#pragma unroll
          for (int kc = 0; kc < 4; ++kc) {
            FragAB Bt;
            loadB(Bt, s_whh,       ct * 16, STRIDE_W, kc * 32, lane); Cr.v = wmma_bf16(A[kc], Bt, Cr.v);
            loadB(Bt, s_whh, 128 + ct * 16, STRIDE_W, kc * 32, lane); Cz.v = wmma_bf16(A[kc], Bt, Cz.v);
            loadB(Bt, s_whh, 256 + ct * 16, STRIDE_W, kc * 32, lane); Cn.v = wmma_bf16(A[kc], Bt, Cn.v);
          }
          const float bir = bihE[h], biz = bihE[128 + h], bin = bihE[256 + h];
          const float bhr = bhhE[h], bhz = bhhE[128 + h], bhn = bhhE[256 + h];
          float wr[3], wz[3], wn[3];
#pragma unroll
          for (int e = 0; e < 3; ++e) {
            wr[e] = WihE[h * 3 + e];
            wz[e] = WihE[(128 + h) * 3 + e];
            wn[e] = WihE[(256 + h) * 3 + e];
          }
#pragma unroll
          for (int r = 0; r < 8; ++r) {
            const int m = r + ((lane >> 4) << 3);
            const int R = row0 + m;
            const int e = et[r];
            const float gir = ((e == 0) ? wr[0] : (e == 1) ? wr[1] : wr[2]) + bir;
            const float giz = ((e == 0) ? wz[0] : (e == 1) ? wz[1] : wz[2]) + biz;
            const float gin = ((e == 0) ? wn[0] : (e == 1) ? wn[1] : wn[2]) + bin;
            const float rg = sigmoidf(gir + Cr.f[r] + bhr);
            const float zg = sigmoidf(giz + Cz.f[r] + bhz);
            const float ng = tanh_fast(gin + rg * (Cn.f[r] + bhn));
            const float hp = bf2f(P.Hst[R * HS + h]);
            const float He = (1.f - zg) * ng + zg * hp;
            strip[m * STRIDE_S + h] = f2bf(He);
          }
        }

        // ---- phase 2 on this strip: gated = sigmoid(inp@Wg+bg)*(inp@Wm) ----
        FragAB A2[4];
#pragma unroll
        for (int kc = 0; kc < 4; ++kc) loadA(A2[kc], strip, 0, STRIDE_S, kc * 32, lane);

        for (int ct2 = 0; ct2 < 8; ++ct2) {
          const int h2 = ct2 * 16 + (lane & 15);
          FragC Cg, Cm;
          Cg.v = (v8f){0,0,0,0,0,0,0,0}; Cm.v = Cg.v;
#pragma unroll
          for (int kc = 0; kc < 4; ++kc) {
            FragAB Bt;
            loadB(Bt, WgB, ct2 * 16, HS, kc * 32, lane); Cg.v = wmma_bf16(A2[kc], Bt, Cg.v);
            loadB(Bt, WmB, ct2 * 16, HS, kc * 32, lane); Cm.v = wmma_bf16(A2[kc], Bt, Cm.v);
          }
          const float bgv = bgF[h2];
          float ge[3], me[3];
#pragma unroll
          for (int e = 0; e < 3; ++e) {
            ge[e] = Wg32[h2 * 131 + 128 + e];   // one-hot edge-type columns
            me[e] = Wm32[h2 * 131 + 128 + e];
          }
#pragma unroll
          for (int r = 0; r < 8; ++r) {
            const float mk = msk[r];
            if (mk != 0.f) {
              const int m = r + ((lane >> 4) << 3);
              const int R = row0 + m, b = R & 31;
              const int e = et[r];
              const float g1 = Cg.f[r] + bgv + ((e == 0) ? ge[0] : (e == 1) ? ge[1] : ge[2]);
              const float m1 = Cm.f[r] +       ((e == 0) ? me[0] : (e == 1) ? me[1] : me[2]);
              atomicAdd(&s_Hmsg[b * HS + h2], sigmoidf(g1) * m1 * mk);   // ds_add_f32
            }
          }
        }
      } // strips
      __syncthreads();

      // -------- Phase 3: H_in = (v==start) ? carry : Hmsg; vertex GRU -------
      unsigned short* s_Hin = &s_strip[0][0];   // bf16 [32][130]
      {
        const int b = tid >> 3, h0 = (tid & 7) << 4;
#pragma unroll
        for (int i = 0; i < 16; ++i) {
          const int h = h0 + i;
          float val = (v == start) ? carry[b * HS + h] : s_Hmsg[b * HS + h];
          s_Hmsg[b * HS + h] = val;             // keep f32 H_in for the z*h term
          s_Hin[b * STRIDE_S + h] = f2bf(val);
        }
      }
      __syncthreads();

      {
        const int ct = wave;                    // wave 0..7 -> col tile ct
        const int h  = ct * 16 + (lane & 15);
        const float bir = bihV[h], biz = bihV[128 + h], bin = bihV[256 + h];
        const float bhr = bhhV[h], bhz = bhhV[128 + h], bhn = bhhV[256 + h];
        for (int rt = 0; rt < 2; ++rt) {
          FragC Cr, Cz, Cn;
          Cr.v = (v8f){0,0,0,0,0,0,0,0}; Cz.v = Cr.v; Cn.v = Cr.v;
#pragma unroll
          for (int kc = 0; kc < 4; ++kc) {
            FragAB Af, Bt;
            loadA(Af, s_Hin, rt * 16, STRIDE_S, kc * 32, lane);
            loadB(Bt, WhhV,       ct * 16, HS, kc * 32, lane); Cr.v = wmma_bf16(Af, Bt, Cr.v);
            loadB(Bt, WhhV, 128 + ct * 16, HS, kc * 32, lane); Cz.v = wmma_bf16(Af, Bt, Cz.v);
            loadB(Bt, WhhV, 256 + ct * 16, HS, kc * 32, lane); Cn.v = wmma_bf16(Af, Bt, Cn.v);
          }
#pragma unroll
          for (int r = 0; r < 8; ++r) {
            const int b  = rt * 16 + r + ((lane >> 4) << 3);
            const int vt = P.v_types[b * Nv + v];
            const float gir = WihV[h * 8 + vt]         + bir;
            const float giz = WihV[(128 + h) * 8 + vt] + biz;
            const float gin = WihV[(256 + h) * 8 + vt] + bin;
            const float rg = sigmoidf(gir + Cr.f[r] + bhr);
            const float zg = sigmoidf(giz + Cz.f[r] + bhz);
            const float ng = tanh_fast(gin + rg * (Cn.f[r] + bhn));
            const float hin = s_Hmsg[b * HS + h];
            const float Hv  = (1.f - zg) * ng + zg * hin;
            P.Hst[(v * 32 + b) * HS + h] = f2bf(Hv);
            if (v == start) carry[b * HS + h] = Hv;
          }
        }
      }
      __threadfence();
      __syncthreads();
    } // steps

    if (pass == 4) {   // snapshot Sf[:, N-1] after the last forward pass
      const int b = tid >> 3, h0 = (tid & 7) << 4;
#pragma unroll
      for (int i = 0; i < 16; ++i) {
        const int h = h0 + i;
        P.SfFin[b * HS + h] = bf2f(P.Hst[((Nv - 1) * 32 + b) * HS + h]);
      }
      __threadfence();
      __syncthreads();
    }
  } // passes

  // -------- classifier: unify + BN + MLP(relu) + BN + linear ----------------
  // After pass 5, P.Hst rows 0..31 hold Sb[:, 0].
  float* uo = s_Hmsg;                       // [32][128]
  float* h1 = (float*)&s_strip[0][0];       // [32][256] (32 KB <= strip area)
  {
    const int b = tid >> 3, h0 = (tid & 7) << 4;
    for (int i = 0; i < 16; ++i) {
      const int h = h0 + i;
      float acc = P.unify_b[h];
      const float* wrow = P.unify_W + h * 256;
      for (int k = 0; k < HS; ++k) acc += P.SfFin[b * HS + k] * wrow[k];
      for (int k = 0; k < HS; ++k) acc += bf2f(P.Hst[b * HS + k]) * wrow[128 + k];
      uo[b * HS + h] = acc;
    }
  }
  __syncthreads();
  if (tid < HS) {
    const int h = tid;
    float mu = 0.f;
    for (int b = 0; b < Bsz; ++b) mu += uo[b * HS + h];
    mu *= (1.f / 32.f);
    float var = 0.f;
    for (int b = 0; b < Bsz; ++b) { float d = uo[b * HS + h] - mu; var += d * d; }
    var *= (1.f / 32.f);
    const float sc = rsqrtf(var + 1e-5f) * P.ubn_g[h];
    const float bb = P.ubn_b[h];
    for (int b = 0; b < Bsz; ++b) uo[b * HS + h] = (uo[b * HS + h] - mu) * sc + bb;
  }
  __syncthreads();
  {
    const int b = tid >> 3, c0 = (tid & 7) << 5;
    for (int i = 0; i < 32; ++i) {
      const int c = c0 + i;
      float acc = P.cls_b1[c];
      const float* wrow = P.cls_W1 + c * HS;
      for (int k = 0; k < HS; ++k) acc += uo[b * HS + k] * wrow[k];
      h1[b * 256 + c] = fmaxf(acc, 0.f);
    }
  }
  __syncthreads();
  if (tid < 256) {
    const int c = tid;
    float mu = 0.f;
    for (int b = 0; b < Bsz; ++b) mu += h1[b * 256 + c];
    mu *= (1.f / 32.f);
    float var = 0.f;
    for (int b = 0; b < Bsz; ++b) { float d = h1[b * 256 + c] - mu; var += d * d; }
    var *= (1.f / 32.f);
    const float sc = rsqrtf(var + 1e-5f) * P.cbn_g[c];
    const float bb = P.cbn_b[c];
    for (int b = 0; b < Bsz; ++b) h1[b * 256 + c] = (h1[b * 256 + c] - mu) * sc + bb;
  }
  __syncthreads();
  if (tid < Bsz) {
    const int b = tid;
    float acc = P.cls_b2[0];
    for (int c = 0; c < 256; ++c) acc += h1[b * 256 + c] * P.cls_W2[c];
    P.out[b] = acc;
  }
}

// ---------------------------------------------------------------------------
extern "C" void kernel_launch(void* const* d_in, const int* in_sizes, int n_in,
                              void* d_out, int out_size, void* d_ws, size_t ws_size,
                              hipStream_t stream) {
  (void)in_sizes; (void)n_in; (void)out_size; (void)ws_size;
  char* ws = (char*)d_ws;

  KParams P;
  P.v_types = (const int*)d_in[0];
  P.e_types = (const int*)d_in[1];
  P.adj     = (const float*)d_in[2];
  P.Wih_fv = (const float*)d_in[3];  P.bih_fv = (const float*)d_in[5];  P.bhh_fv = (const float*)d_in[6];
  P.Wih_bv = (const float*)d_in[7];  P.bih_bv = (const float*)d_in[9];  P.bhh_bv = (const float*)d_in[10];
  P.Wih_fe = (const float*)d_in[11]; P.bih_fe = (const float*)d_in[13]; P.bhh_fe = (const float*)d_in[14];
  P.Wih_be = (const float*)d_in[15]; P.bih_be = (const float*)d_in[17]; P.bhh_be = (const float*)d_in[18];
  P.gate_f_W = (const float*)d_in[19]; P.gate_f_b = (const float*)d_in[20]; P.map_f_W = (const float*)d_in[21];
  P.gate_b_W = (const float*)d_in[22]; P.gate_b_b = (const float*)d_in[23]; P.map_b_W = (const float*)d_in[24];
  P.unify_W = (const float*)d_in[25]; P.unify_b = (const float*)d_in[26];
  P.ubn_g = (const float*)d_in[27];   P.ubn_b = (const float*)d_in[28];
  P.cls_W1 = (const float*)d_in[29];  P.cls_b1 = (const float*)d_in[30];
  P.cbn_g = (const float*)d_in[31];   P.cbn_b = (const float*)d_in[32];
  P.cls_W2 = (const float*)d_in[33];  P.cls_b2 = (const float*)d_in[34];

  P.wb     = (const unsigned short*)(ws + WS_WB);
  P.Hst    = (unsigned short*)(ws + WS_HST);
  P.carryF = (float*)(ws + WS_CARRYF);
  P.carryB = (float*)(ws + WS_CARRYB);
  P.SfFin  = (float*)(ws + WS_SFFIN);
  P.out    = (float*)d_out;

  dvae_prep_kernel<<<(WB_ELEMS + 255) / 256, 256, 0, stream>>>(
      (const float*)d_in[12], (const float*)d_in[16],   // Whh_fe, Whh_be
      (const float*)d_in[4],  (const float*)d_in[8],    // Whh_fv, Whh_bv
      (const float*)d_in[19], (const float*)d_in[21],   // gate_f_W, map_f_W
      (const float*)d_in[22], (const float*)d_in[24],   // gate_b_W, map_b_W
      (unsigned short*)(ws + WS_WB));

  dvae_persistent_kernel<<<1, 256, 0, stream>>>(P);
}
